// SocialEncoder_55594056680054
// MI455X (gfx1250) — matrix operand order repaired and checked by reading, not compiled, not run on hardware
//
#include <hip/hip_runtime.h>
#include <hip/hip_bf16.h>

typedef __attribute__((ext_vector_type(16))) _Float16 v16h;
typedef __attribute__((ext_vector_type(8)))  _Float16 v8h;
typedef __attribute__((ext_vector_type(8)))  float    v8f;

#define WMMA_F16(a, b, c) \
  __builtin_amdgcn_wmma_f32_16x16x32_f16(false, (a), false, (b), (short)0, (c), false, false)

#define Bdim   8
#define Sdim   1024
#define Hdim   1024
#define NHdim  16
#define HDdim  64
#define FFdim  4096
#define MAXREL 50

// ---------------- fragment loaders (layouts per CDNA5 ISA 7.12.2) ----------------

// A-matrix 16x32 f16: half e -> K = (e&7) + 16*(e>>3) + 8*hi, row M = lane&15
// per-lane reads are two contiguous 8-half runs -> 2x ds_load_b128
__device__ inline v16h load_frag_a(const _Float16* base, int row0, int k0, int ld) {
  const int lane = threadIdx.x & 31;
  const int hi = lane >> 4, ml = lane & 15;
  const _Float16* p = base + (size_t)(row0 + ml) * ld + k0;
  v16h a;
#pragma unroll
  for (int e = 0; e < 16; ++e) {
    int k = (e & 7) + ((e >> 3) << 4) + (hi << 3);
    a[e] = p[k];
  }
  return a;
}

// B = X^T where X is row-major [rows, k]: B[k][n] = X[row0+n][k0+k]
// half e -> K = (e&7) + 16*(e>>3) + 8*hi (same order as A); two 8-half runs per lane
__device__ inline v16h load_frag_bt(const _Float16* base, int row0, int k0, int ld) {
  const int lane = threadIdx.x & 31;
  const int hi = lane >> 4, ml = lane & 15;
  const _Float16* p = base + (size_t)(row0 + ml) * ld + k0;
  v16h b;
#pragma unroll
  for (int e = 0; e < 16; ++e) b[e] = p[(e & 7) + ((e >> 3) << 4) + (hi << 3)];
  return b;
}

// ---------------- converts ----------------
__global__ __launch_bounds__(256) void cvt_kernel(const float* __restrict__ in,
                                                  _Float16* __restrict__ out, int n) {
  int i = blockIdx.x * 256 + threadIdx.x;
  if (i < n) out[i] = (_Float16)in[i];
}

// in [K][N] f32 -> out [N][K] f16 (transposed weights so GEMM B-tiles load like A-tiles)
__global__ __launch_bounds__(256) void cvtT_kernel(const float* __restrict__ in,
                                                   _Float16* __restrict__ out, int K, int N) {
  __shared__ _Float16 tile[32][33];
  const int k0 = blockIdx.y * 32, n0 = blockIdx.x * 32;
  const int r = threadIdx.x >> 3, c4 = (threadIdx.x & 7) * 4;
#pragma unroll
  for (int j = 0; j < 4; ++j)
    tile[r][c4 + j] = (_Float16)in[(size_t)(k0 + r) * N + n0 + c4 + j];
  __syncthreads();
#pragma unroll
  for (int j = 0; j < 4; ++j)
    out[(size_t)(n0 + r) * K + k0 + c4 + j] = tile[c4 + j][r];
}

// rel_bias [50,16,64] -> table [50,16] (sum over head_dim)
__global__ __launch_bounds__(256) void relbias_kernel(const float* __restrict__ rel,
                                                      float* __restrict__ tab) {
  int i = blockIdx.x * 256 + threadIdx.x;
  if (i < MAXREL * NHdim) {
    float s = 0.f;
#pragma unroll 8
    for (int d = 0; d < HDdim; ++d) s += rel[(size_t)i * HDdim + d];
    tab[i] = s;
  }
}

// ---------------- WMMA GEMM: C[M,N] = A[M,K] @ Bt[N,K]^T + bias ----------------
// 128x128 block tile, 8 waves of 32x64, K-step 32, double-buffered LDS.
template <int FLAGS>  // bit0 = relu, bit1 = f16 output
__global__ __launch_bounds__(256) void gemm_kernel(
    const _Float16* __restrict__ A, const _Float16* __restrict__ Bt,
    const float* __restrict__ bias, float* __restrict__ Cf, _Float16* __restrict__ Ch,
    int M, int N, int K)
{
  __shared__ __attribute__((aligned(16))) _Float16 As[2][128 * 32];
  __shared__ __attribute__((aligned(16))) _Float16 Bs[2][128 * 32];
  const int tid = threadIdx.x;
  const int wave = tid >> 5, lane = tid & 31;
  const int hi = lane >> 4, ml = lane & 15;
  const int wm = wave & 3, wn = wave >> 2;
  const int blockM = blockIdx.y * 128;
  const int blockN = blockIdx.x * 128;

  const int lr = tid >> 2;           // 0..63
  const int lc = (tid & 3) * 8;      // 0,8,16,24
  const _Float16* Ap0 = A  + (size_t)(blockM + lr) * K + lc;
  const _Float16* Ap1 = A  + (size_t)(blockM + lr + 64) * K + lc;
  const _Float16* Bp0 = Bt + (size_t)(blockN + lr) * K + lc;
  const _Float16* Bp1 = Bt + (size_t)(blockN + lr + 64) * K + lc;

  v8f zero = {};
  v8f acc[2][4];
#pragma unroll
  for (int mi = 0; mi < 2; ++mi)
#pragma unroll
    for (int ni = 0; ni < 4; ++ni) acc[mi][ni] = zero;

  v8h ra0, ra1, rb0, rb1;
  auto gload = [&](int k0) {
    ra0 = *(const v8h*)(Ap0 + k0);
    ra1 = *(const v8h*)(Ap1 + k0);
    rb0 = *(const v8h*)(Bp0 + k0);
    rb1 = *(const v8h*)(Bp1 + k0);
  };
  auto sstore = [&](int buf) {
    *(v8h*)(&As[buf][lr * 32 + lc])        = ra0;
    *(v8h*)(&As[buf][(lr + 64) * 32 + lc]) = ra1;
    *(v8h*)(&Bs[buf][lr * 32 + lc])        = rb0;
    *(v8h*)(&Bs[buf][(lr + 64) * 32 + lc]) = rb1;
  };
  auto compute = [&](int buf) {
    v16h af[2], bf[4];
#pragma unroll
    for (int mi = 0; mi < 2; ++mi) af[mi] = load_frag_a(&As[buf][0], wm * 32 + mi * 16, 0, 32);
#pragma unroll
    for (int ni = 0; ni < 4; ++ni) bf[ni] = load_frag_bt(&Bs[buf][0], wn * 64 + ni * 16, 0, 32);
#pragma unroll
    for (int mi = 0; mi < 2; ++mi)
#pragma unroll
      for (int ni = 0; ni < 4; ++ni) acc[mi][ni] = WMMA_F16(af[mi], bf[ni], acc[mi][ni]);
  };

  gload(0);
  sstore(0);
  int buf = 0;
  for (int k0 = 32; k0 < K; k0 += 32) {
    __syncthreads();
    gload(k0);        // prefetch next tile into registers
    compute(buf);     // WMMAs on current LDS buffer
    sstore(buf ^ 1);  // fill the other buffer
    buf ^= 1;
  }
  __syncthreads();
  compute(buf);

#pragma unroll
  for (int mi = 0; mi < 2; ++mi)
#pragma unroll
    for (int ni = 0; ni < 4; ++ni)
#pragma unroll
      for (int r = 0; r < 8; ++r) {
        int row = blockM + wm * 32 + mi * 16 + hi * 8 + r;
        int col = blockN + wn * 64 + ni * 16 + ml;
        float v = acc[mi][ni][r] + bias[col];
        if (FLAGS & 1) v = fmaxf(v, 0.f);
        size_t o = (size_t)row * N + col;
        if (FLAGS & 2) Ch[o] = (_Float16)v;
        else           Cf[o] = v;
      }
}

// ---------------- flash attention: one block per (b, h, 128-row tile) ----------------
__global__ __launch_bounds__(256) void attn_kernel(
    const _Float16* __restrict__ Q, const _Float16* __restrict__ Km,
    const _Float16* __restrict__ V, const float* __restrict__ btab,
    const float* __restrict__ scaling, float* __restrict__ Out)
{
  __shared__ __attribute__((aligned(16))) _Float16 Qs[128 * 64];
  __shared__ __attribute__((aligned(16))) _Float16 Ks[128 * 64];
  __shared__ __attribute__((aligned(16))) _Float16 Vt[64 * 136];   // V transposed [d][j], padded
  __shared__ __attribute__((aligned(16))) _Float16 Ps[8][16 * 128];

  const int tid = threadIdx.x;
  const int wave = tid >> 5, lane = tid & 31;
  const int hi = lane >> 4, ml = lane & 15;
  const int h = blockIdx.y, b = blockIdx.z, ib = blockIdx.x;
  const size_t base = ((size_t)b * Sdim) * Hdim + h * HDdim;

  for (int t = tid; t < 1024; t += 256) {  // Q block 128x64
    int r = t >> 3, c = t & 7;
    *(v8h*)(Qs + r * 64 + c * 8) =
        *(const v8h*)(Q + base + (size_t)(ib * 128 + r) * Hdim + c * 8);
  }

  const float sc = scaling[h];
  float mrow[8], lrow[8];
#pragma unroll
  for (int r = 0; r < 8; ++r) { mrow[r] = -1e30f; lrow[r] = 0.f; }
  v8f zero = {};
  v8f O[4];
#pragma unroll
  for (int dt = 0; dt < 4; ++dt) O[dt] = zero;
  const int i0 = ib * 128 + wave * 16;

  for (int jb = 0; jb < 8; ++jb) {
    __syncthreads();
    for (int t = tid; t < 1024; t += 256) {  // K row-major; V transposed into Vt
      int r = t >> 3, c = t & 7;
      *(v8h*)(Ks + r * 64 + c * 8) =
          *(const v8h*)(Km + base + (size_t)(jb * 128 + r) * Hdim + c * 8);
      v8h vv = *(const v8h*)(V + base + (size_t)(jb * 128 + r) * Hdim + c * 8);
#pragma unroll
      for (int e = 0; e < 8; ++e) Vt[(size_t)(c * 8 + e) * 136 + r] = vv[e];
    }
    __syncthreads();

    // scores: 16 rows x 128 cols for this wave (Q frags hoisted out of jj loop)
    v16h aq0 = load_frag_a(Qs, wave * 16, 0, 64);
    v16h aq1 = load_frag_a(Qs, wave * 16, 32, 64);
    v8f sf[8];
#pragma unroll
    for (int jj = 0; jj < 8; ++jj) {
      v8f s = zero;
      s = WMMA_F16(aq0, load_frag_bt(Ks, jj * 16, 0, 64), s);
      s = WMMA_F16(aq1, load_frag_bt(Ks, jj * 16, 32, 64), s);
#pragma unroll
      for (int r = 0; r < 8; ++r) {
        int gi = i0 + hi * 8 + r;
        int gj = jb * 128 + jj * 16 + ml;
        int d = gi - gj; if (d < 0) d = -d; if (d > MAXREL - 1) d = MAXREL - 1;
        s[r] = s[r] * sc + btab[d * NHdim + h];
      }
      sf[jj] = s;
    }

    // online softmax (row m = hi*8 + r lives in register r across 16 lanes)
    float rmax[8];
#pragma unroll
    for (int r = 0; r < 8; ++r) {
      float m = sf[0][r];
#pragma unroll
      for (int jj = 1; jj < 8; ++jj) m = fmaxf(m, sf[jj][r]);
      rmax[r] = m;
    }
#pragma unroll
    for (int off = 8; off >= 1; off >>= 1)
#pragma unroll
      for (int r = 0; r < 8; ++r) rmax[r] = fmaxf(rmax[r], __shfl_xor(rmax[r], off, 32));

    float scl[8], rsum[8];
#pragma unroll
    for (int r = 0; r < 8; ++r) {
      float nm = fmaxf(mrow[r], rmax[r]);
      scl[r] = __expf(mrow[r] - nm);
      mrow[r] = nm;
      rsum[r] = 0.f;
    }
#pragma unroll
    for (int jj = 0; jj < 8; ++jj)
#pragma unroll
      for (int r = 0; r < 8; ++r) {
        float p = __expf(sf[jj][r] - mrow[r]);
        rsum[r] += p;
        Ps[wave][(hi * 8 + r) * 128 + jj * 16 + ml] = (_Float16)p;
      }
#pragma unroll
    for (int off = 8; off >= 1; off >>= 1)
#pragma unroll
      for (int r = 0; r < 8; ++r) rsum[r] += __shfl_xor(rsum[r], off, 32);
#pragma unroll
    for (int r = 0; r < 8; ++r) lrow[r] = lrow[r] * scl[r] + rsum[r];
#pragma unroll
    for (int dt = 0; dt < 4; ++dt)
#pragma unroll
      for (int r = 0; r < 8; ++r) O[dt][r] *= scl[r];

    // O += P @ V  via transposed Vt (all-b128 fragment loads)
#pragma unroll
    for (int kk = 0; kk < 4; ++kk) {
      v16h a = load_frag_a(Ps[wave], 0, kk * 32, 128);
#pragma unroll
      for (int dt = 0; dt < 4; ++dt) {
        v16h bv = load_frag_bt(Vt, dt * 16, kk * 32, 136);
        O[dt] = WMMA_F16(a, bv, O[dt]);
      }
    }
  }

#pragma unroll
  for (int dt = 0; dt < 4; ++dt)
#pragma unroll
    for (int r = 0; r < 8; ++r) {
      float v = O[dt][r] / lrow[r];
      size_t row = (size_t)(b * Sdim + i0 + hi * 8 + r);
      Out[row * Hdim + h * HDdim + dt * 16 + ml] = v;
    }
}

// ---------------- layernorm helpers/kernels ----------------
__device__ inline float block_sum(float v, float* red) {
#pragma unroll
  for (int off = 16; off >= 1; off >>= 1) v += __shfl_xor(v, off, 32);
  __syncthreads();
  if ((threadIdx.x & 31) == 0) red[threadIdx.x >> 5] = v;
  __syncthreads();
  float s = 0.f;
#pragma unroll
  for (int i = 0; i < 8; ++i) s += red[i];
  return s;
}

__global__ __launch_bounds__(256) void ln1_kernel(
    const float* __restrict__ x, const float* __restrict__ attn,
    const float* __restrict__ rv, const float* __restrict__ g,
    const float* __restrict__ be, float* __restrict__ x1f, _Float16* __restrict__ x1h)
{
  __shared__ float red[8];
  const size_t off = (size_t)blockIdx.x * Hdim;
  float t[4]; float s = 0.f;
#pragma unroll
  for (int j = 0; j < 4; ++j) {
    int c = threadIdx.x + j * 256;
    t[j] = x[off + c] + attn[off + c] + rv[off + c];
    s += t[j];
  }
  float mean = block_sum(s, red) * (1.f / Hdim);
  float vs = 0.f;
#pragma unroll
  for (int j = 0; j < 4; ++j) { float d = t[j] - mean; vs += d * d; }
  float inv = rsqrtf(block_sum(vs, red) * (1.f / Hdim) + 1e-5f);
#pragma unroll
  for (int j = 0; j < 4; ++j) {
    int c = threadIdx.x + j * 256;
    float y = (t[j] - mean) * inv * g[c] + be[c];
    x1f[off + c] = y;
    x1h[off + c] = (_Float16)y;
  }
}

__global__ __launch_bounds__(256) void final_kernel(
    const float* __restrict__ ffpre, const float* __restrict__ x1f,
    const float* __restrict__ fg, const float* __restrict__ fb,
    const float* __restrict__ g2, const float* __restrict__ b2, float* __restrict__ out)
{
  __shared__ float red[8];
  const size_t off = (size_t)blockIdx.x * Hdim;
  float t[4]; float s = 0.f;
#pragma unroll
  for (int j = 0; j < 4; ++j) {
    int c = threadIdx.x + j * 256;
    t[j] = ffpre[off + c];
    s += t[j];
  }
  float mean = block_sum(s, red) * (1.f / Hdim);
  float vs = 0.f;
#pragma unroll
  for (int j = 0; j < 4; ++j) { float d = t[j] - mean; vs += d * d; }
  float inv = rsqrtf(block_sum(vs, red) * (1.f / Hdim) + 1e-5f);

  float u[4]; s = 0.f;
#pragma unroll
  for (int j = 0; j < 4; ++j) {
    int c = threadIdx.x + j * 256;
    float ff = (t[j] - mean) * inv * fg[c] + fb[c];
    u[j] = x1f[off + c] + ff;
    s += u[j];
  }
  mean = block_sum(s, red) * (1.f / Hdim);
  vs = 0.f;
#pragma unroll
  for (int j = 0; j < 4; ++j) { float d = u[j] - mean; vs += d * d; }
  inv = rsqrtf(block_sum(vs, red) * (1.f / Hdim) + 1e-5f);
#pragma unroll
  for (int j = 0; j < 4; ++j) {
    int c = threadIdx.x + j * 256;
    out[off + c] = (u[j] - mean) * inv * g2[c] + b2[c];
  }
}

// ---------------- host orchestration ----------------
extern "C" void kernel_launch(void* const* d_in, const int* in_sizes, int n_in,
                              void* d_out, int out_size, void* d_ws, size_t ws_size,
                              hipStream_t stream) {
  const float* x     = (const float*)d_in[0];
  const float* lq_w  = (const float*)d_in[1];  const float* lq_b  = (const float*)d_in[2];
  const float* lk_w  = (const float*)d_in[3];  const float* lk_b  = (const float*)d_in[4];
  const float* lv_w  = (const float*)d_in[5];  const float* lv_b  = (const float*)d_in[6];
  const float* aq_w  = (const float*)d_in[7];  const float* aq_b  = (const float*)d_in[8];
  const float* ak_w  = (const float*)d_in[9];  const float* ak_b  = (const float*)d_in[10];
  const float* av_w  = (const float*)d_in[11]; const float* av_b  = (const float*)d_in[12];
  const float* arv_w = (const float*)d_in[13]; const float* arv_b = (const float*)d_in[14];
  const float* relb  = (const float*)d_in[15];
  const float* scal  = (const float*)d_in[16];
  const float* ff1_w = (const float*)d_in[17]; const float* ff1_b = (const float*)d_in[18];
  const float* ff2_w = (const float*)d_in[19]; const float* ff2_b = (const float*)d_in[20];
  const float* ffn_g = (const float*)d_in[21]; const float* ffn_b = (const float*)d_in[22];
  const float* ln1_g = (const float*)d_in[23]; const float* ln1_b = (const float*)d_in[24];
  const float* ln2_g = (const float*)d_in[25]; const float* ln2_b = (const float*)d_in[26];
  float* out = (float*)d_out;

  char* ws = (char*)d_ws;
  size_t off = 0;
  auto alloc = [&](size_t bytes) -> void* {
    off = (off + 255) & ~(size_t)255;
    void* p = ws + off;
    off += bytes;
    return p;
  };
  const size_t MS = (size_t)Bdim * Sdim;  // 8192 rows

  _Float16* xh   = (_Float16*)alloc(MS * Hdim * 2);
  _Float16* w_lq = (_Float16*)alloc((size_t)Hdim * Hdim * 2);   // all weights stored transposed [N][K]
  _Float16* w_lk = (_Float16*)alloc((size_t)Hdim * Hdim * 2);
  _Float16* w_lv = (_Float16*)alloc((size_t)Hdim * Hdim * 2);
  _Float16* w_aq = (_Float16*)alloc((size_t)Hdim * Hdim * 2);
  _Float16* w_ak = (_Float16*)alloc((size_t)Hdim * Hdim * 2);
  _Float16* w_av = (_Float16*)alloc((size_t)Hdim * Hdim * 2);
  _Float16* w_rv = (_Float16*)alloc((size_t)Hdim * Hdim * 2);
  _Float16* w_f1 = (_Float16*)alloc((size_t)Hdim * FFdim * 2);
  _Float16* w_f2 = (_Float16*)alloc((size_t)FFdim * Hdim * 2);
  float*    btab = (float*)alloc(MAXREL * NHdim * 4);

  _Float16* qkv = (_Float16*)alloc(6 * MS * Hdim * 2);  // q,k,v,Q,K,V contiguous
  _Float16* q16 = qkv;
  _Float16* k16 = qkv + 1 * MS * Hdim;
  _Float16* v16 = qkv + 2 * MS * Hdim;
  _Float16* Qp  = qkv + 3 * MS * Hdim;
  _Float16* Kp  = qkv + 4 * MS * Hdim;
  _Float16* Vp  = qkv + 5 * MS * Hdim;
  float* rvf   = (float*)alloc(MS * Hdim * 4);
  float* attnf = (float*)alloc(MS * Hdim * 4);
  float* x1f   = (float*)alloc(MS * Hdim * 4);
  // reuse (stream-ordered, producers finish before overwrite):
  _Float16* x1h   = xh;   // xh dead after q/k/v GEMMs
  _Float16* hmid  = qkv;  // 64MB of the 96MB qkv block; dead after attention
  float*    ffpre = rvf;  // rvf consumed by ln1

  cvt_kernel<<<dim3((unsigned)((MS * Hdim + 255) / 256)), 256, 0, stream>>>(x, xh, (int)(MS * Hdim));
  auto cvtT = [&](const float* in, _Float16* o, int K, int N) {
    cvtT_kernel<<<dim3(N / 32, K / 32), 256, 0, stream>>>(in, o, K, N);
  };
  cvtT(lq_w, w_lq, Hdim, Hdim);  cvtT(lk_w, w_lk, Hdim, Hdim);
  cvtT(lv_w, w_lv, Hdim, Hdim);  cvtT(aq_w, w_aq, Hdim, Hdim);
  cvtT(ak_w, w_ak, Hdim, Hdim);  cvtT(av_w, w_av, Hdim, Hdim);
  cvtT(arv_w, w_rv, Hdim, Hdim);
  cvtT(ff1_w, w_f1, Hdim, FFdim);
  cvtT(ff2_w, w_f2, FFdim, Hdim);
  relbias_kernel<<<dim3(4), 256, 0, stream>>>(relb, btab);

  auto gemm = [&](const _Float16* A, const _Float16* Bt, const float* bias, void* C,
                  int M, int N, int K, int flags) {
    dim3 g(N / 128, M / 128);
    if (flags == 3)
      gemm_kernel<3><<<g, 256, 0, stream>>>(A, Bt, bias, (float*)C, (_Float16*)C, M, N, K);
    else if (flags == 2)
      gemm_kernel<2><<<g, 256, 0, stream>>>(A, Bt, bias, (float*)C, (_Float16*)C, M, N, K);
    else
      gemm_kernel<0><<<g, 256, 0, stream>>>(A, Bt, bias, (float*)C, (_Float16*)C, M, N, K);
  };

  // encoder input projections
  gemm(xh, w_lq, lq_b, q16, (int)MS, Hdim, Hdim, /*f16*/2);
  gemm(xh, w_lk, lk_b, k16, (int)MS, Hdim, Hdim, 2);
  gemm(xh, w_lv, lv_b, v16, (int)MS, Hdim, Hdim, 2);
  // attention projections
  gemm(q16, w_aq, aq_b, Qp, (int)MS, Hdim, Hdim, 2);
  gemm(k16, w_ak, ak_b, Kp, (int)MS, Hdim, Hdim, 2);
  gemm(v16, w_av, av_b, Vp, (int)MS, Hdim, Hdim, 2);
  gemm(v16, w_rv, arv_b, rvf, (int)MS, Hdim, Hdim, /*f32*/0);

  // attention
  attn_kernel<<<dim3(Sdim / 128, NHdim, Bdim), 256, 0, stream>>>(Qp, Kp, Vp, btab, scal, attnf);

  // x1 = LN(x + attn + rv)
  ln1_kernel<<<dim3((unsigned)MS), 256, 0, stream>>>(x, attnf, rvf, ln1_g, ln1_b, x1f, x1h);

  // feed-forward
  gemm(x1h, w_f1, ff1_b, hmid, (int)MS, FFdim, Hdim, /*relu|f16*/3);
  gemm(hmid, w_f2, ff2_b, ffpre, (int)MS, Hdim, FFdim, 0);

  // out = LN(x1 + LN(ffpre))
  final_kernel<<<dim3((unsigned)MS), 256, 0, stream>>>(ffpre, x1f, ffn_g, ffn_b, ln2_g, ln2_b, out);
}
